// TCGNNCell_11647951306947
// MI455X (gfx1250) — compile-verified, matrix operand order checked by reading, not run on hardware
//
#include <hip/hip_runtime.h>

// ---------------------------------------------------------------------------
// TCGNN GRU cell for gfx1250 (MI455X): bf16 WMMA pipeline, f32 accumulate.
//   N=2048 nodes, U=64, DI=1 (I=65), B=128 batch.
//   value = sigmoid((adj @ x) @ W0); c = tanh((adj @ x') @ W1)
//   new_state = u*state + (1-u)*c
// Graph-conv GEMMs use TDM (tensor_load_to_lds) double-buffered K-panels.
// ---------------------------------------------------------------------------

typedef __bf16 bf16_t;
typedef bf16_t v16bf __attribute__((ext_vector_type(16)));
typedef bf16_t v8bf  __attribute__((ext_vector_type(8)));
typedef bf16_t v4bf  __attribute__((ext_vector_type(4)));
typedef float  v8f   __attribute__((ext_vector_type(8)));
typedef float  v4f   __attribute__((ext_vector_type(4)));
typedef unsigned int u32;
typedef u32 v4u __attribute__((ext_vector_type(4)));
typedef int  v8i __attribute__((ext_vector_type(8)));
typedef int  v4i __attribute__((ext_vector_type(4)));

#define NN    2048          // nodes
#define UU    64            // gru units
#define BB    128           // batch
#define II    65            // DI + U
#define NC    (II * BB)     // 8320 packed feature columns
#define KPAD  96            // I padded to 3 x K32 for wmma
#define NUSZ  ((size_t)NN * UU)   // 131072

// workspace layout (bytes)
#define OFF_ADJB 0u                                     // bf16 [2048][2048]
#define OFF_XPT  (OFF_ADJB + (size_t)NN * NN * 2)       // bf16 [8320][2048]
#define OFF_C    (OFF_XPT + (size_t)NC * NN * 2)        // bf16 [2048][8320]
#define OFF_U    (OFF_C   + (size_t)NN * NC * 2)        // bf16 [128][131072]
#define OFF_RS   (OFF_U   + (size_t)BB * NUSZ * 2)      // bf16 [128][131072]

#if defined(__HIP_DEVICE_COMPILE__) && __has_builtin(__builtin_amdgcn_tensor_load_to_lds)
#define USE_TDM 1
#else
#define USE_TDM 0
#endif

// --- WMMA fragment load: 16x32 bf16 tile, K contiguous in LDS rows ---------
// ISA 7.12.2 16-bit A/B layout: lane = r + 16*h holds row r, K = 8h..8h+7 in
// regs 0-3 and K = 16+8h..23+8h in regs 4-7.
__device__ __forceinline__ v16bf load_frag(const bf16_t* rowptr, int kc, int h) {
  const v8bf lo = *(const v8bf*)(rowptr + kc + 8 * h);
  const v8bf hi = *(const v8bf*)(rowptr + kc + 16 + 8 * h);
  v16bf f;
#pragma unroll
  for (int j = 0; j < 8; ++j) { f[j] = lo[j]; f[8 + j] = hi[j]; }
  return f;
}

__device__ __forceinline__ float sigmoid_f(float x) {
  return 1.0f / (1.0f + __expf(-x));
}
__device__ __forceinline__ float tanh_f(float x) {
  float e = __expf(2.0f * x);
  return 1.0f - 2.0f / (e + 1.0f);   // saturates correctly at +/-inf
}

#if USE_TDM
// --- Tensor Data Mover: 2-D bf16 tile (rows x cols) global->LDS ------------
// D# per CDNA5 ISA: tensor dims == tile dims (never OOB),
// row pitch = row_stride_elems, packed destination in LDS.
__device__ __forceinline__ void tdm_load_2d(unsigned lds_off, const void* gptr,
                                            unsigned rows, unsigned cols,
                                            unsigned row_stride_elems) {
  unsigned long long ga = (unsigned long long)(uintptr_t)gptr;
  v4u g0;
  g0[0] = 1u;                                            // count=1 (valid, user)
  g0[1] = lds_off;                                       // lds_addr (bytes)
  g0[2] = (u32)(ga & 0xFFFFFFFFu);                       // global_addr[31:0]
  g0[3] = (u32)((ga >> 32) & 0x01FFFFFFu) | (2u << 30);  // addr[56:32] | type=2
  v8i g1;
  g1[0] = (int)(1u << 16);                               // data_size=1 -> 2 bytes
  g1[1] = (int)((cols & 0xFFFFu) << 16);                 // tensor_dim0[15:0]
  g1[2] = (int)(((cols >> 16) & 0xFFFFu) | ((rows & 0xFFFFu) << 16));  // dim0 hi | dim1 lo
  g1[3] = (int)(((rows >> 16) & 0xFFFFu) | ((cols & 0xFFFFu) << 16));  // dim1 hi | tile_dim0
  g1[4] = (int)(rows & 0xFFFFu);                         // tile_dim1 (tile_dim2=0)
  g1[5] = (int)row_stride_elems;                         // tensor_dim0_stride lo32
  g1[6] = 0;                                             // stride hi | dim1_stride lo
  g1[7] = 0;
  v4i z4 = {0, 0, 0, 0};
#if defined(__clang_major__) && (__clang_major__ >= 23)
  v8i z8 = {0, 0, 0, 0, 0, 0, 0, 0};
  __builtin_amdgcn_tensor_load_to_lds(g0, g1, z4, z4, z8, 0);
#else
  __builtin_amdgcn_tensor_load_to_lds(g0, g1, z4, z4, 0);
#endif
}
#endif  // USE_TDM

// --- pack kernels ----------------------------------------------------------
__global__ __launch_bounds__(256) void pack_adj_kernel(
    const float* __restrict__ adj, bf16_t* __restrict__ adjb) {
  size_t i4 = ((size_t)blockIdx.x * 256 + threadIdx.x) * 4;
  v4f v = *(const v4f*)(adj + i4);
  v4bf o;
#pragma unroll
  for (int j = 0; j < 4; ++j) o[j] = (bf16_t)v[j];
  *(v4bf*)(adjb + i4) = o;
}

// XpT[(i>0 ? i*128+b : b)][m] = x[b, m, i]; src = f32 state
__global__ __launch_bounds__(256) void pack_x_f32_kernel(
    const float* __restrict__ inputs, const float* __restrict__ state,
    bf16_t* __restrict__ xpt) {
  __shared__ float Ts[128 * 65];                 // [m_local][i] padded row 65
  const int t = threadIdx.x;
  const int m0 = blockIdx.x * 128;
  const int b = blockIdx.y;
  for (int idx = t; idx < 128 * 64; idx += 256) {
    int ml = idx >> 6, i = idx & 63;
    Ts[ml * 65 + i] = state[(size_t)b * NUSZ + (size_t)(m0 + ml) * UU + i];
  }
  __syncthreads();
  for (int e = t; e < 64 * 128; e += 256) {
    int i = e >> 7, ml = e & 127;
    xpt[(size_t)((i + 1) * BB + b) * NN + m0 + ml] = (bf16_t)Ts[ml * 65 + i];
  }
  if (t < 128)
    xpt[(size_t)b * NN + m0 + t] = (bf16_t)inputs[(size_t)b * NN + m0 + t];
}

// same, but src = bf16 r*state; input column (i=0) left untouched
__global__ __launch_bounds__(256) void pack_x_bf16_kernel(
    const bf16_t* __restrict__ rs, bf16_t* __restrict__ xpt) {
  __shared__ bf16_t Ts[128 * 65];
  const int t = threadIdx.x;
  const int m0 = blockIdx.x * 128;
  const int b = blockIdx.y;
  for (int idx = t; idx < 128 * 64; idx += 256) {
    int ml = idx >> 6, i = idx & 63;
    Ts[ml * 65 + i] = rs[(size_t)b * NUSZ + (size_t)(m0 + ml) * UU + i];
  }
  __syncthreads();
  for (int e = t; e < 64 * 128; e += 256) {
    int i = e >> 7, ml = e & 127;
    xpt[(size_t)((i + 1) * BB + b) * NN + m0 + ml] = Ts[ml * 65 + i];
  }
}

// --- shared compute: one 128x128 tile-panel of wmma ------------------------
__device__ __forceinline__ void gemm_compute_panel(
    const bf16_t* __restrict__ As, const bf16_t* __restrict__ Bs,
    int wave, int r, int h, v8f* acc) {
#pragma unroll
  for (int kc = 0; kc < 64; kc += 32) {
    const v16bf a = load_frag(As + (wave * 16 + r) * 64, kc, h);
    v16bf bfr[8];
#pragma unroll
    for (int tt = 0; tt < 8; ++tt)
      bfr[tt] = load_frag(Bs + (tt * 16 + r) * 64, kc, h);
#pragma unroll
    for (int tt = 0; tt < 8; ++tt)
      acc[tt] = __builtin_amdgcn_wmma_f32_16x16x32_bf16(
          false, a, false, bfr[tt], (short)0, acc[tt], false, false);
  }
}

// --- big graph-conv GEMM: C[n][c] = sum_m adjb[n][m] * XpT[c][m] -----------
// block = 128n x 128c tile, 8 waves, double-buffered 128x64 K-panels.
__global__ __launch_bounds__(256) void gemm_adj_kernel(
    const bf16_t* __restrict__ adjb, const bf16_t* __restrict__ xpt,
    bf16_t* __restrict__ cout) {
  __shared__ __align__(16) bf16_t As[2][128 * 64];
  __shared__ __align__(16) bf16_t Bs[2][128 * 64];
  const int t = threadIdx.x;
  const int wave = t >> 5, lane = t & 31;
  const int r = lane & 15, h = lane >> 4;
  const int c0 = blockIdx.x * 128;
  const int n0 = blockIdx.y * 128;
  const int NT = NN / 64;

  v8f acc[8] = {};

#if USE_TDM
  // prologue: wave 0 kicks DMA of panel 0 into buffer 0
  if (t < 32) {
    tdm_load_2d((u32)(uintptr_t)(void*)&As[0][0], adjb + (size_t)n0 * NN,
                128, 64, NN);
    tdm_load_2d((u32)(uintptr_t)(void*)&Bs[0][0], xpt + (size_t)c0 * NN,
                128, 64, NN);
  }
  for (int kt = 0; kt < NT; ++kt) {
    const int buf = kt & 1;
    __syncthreads();                 // all waves done with buf^1's old panel
    if (t < 32) {
      if (kt + 1 < NT) {             // DMA panel kt+1 while kt computes
        const int kn = (kt + 1) * 64;
        tdm_load_2d((u32)(uintptr_t)(void*)&As[buf ^ 1][0],
                    adjb + (size_t)n0 * NN + kn, 128, 64, NN);
        tdm_load_2d((u32)(uintptr_t)(void*)&Bs[buf ^ 1][0],
                    xpt + (size_t)c0 * NN + kn, 128, 64, NN);
        __builtin_amdgcn_s_wait_tensorcnt(2);   // panel kt complete
      } else {
        __builtin_amdgcn_s_wait_tensorcnt(0);
      }
    }
    __syncthreads();                 // panel kt visible to all waves
    gemm_compute_panel(&As[buf][0], &Bs[buf][0], wave, r, h, acc);
  }
#else
  for (int kt = 0; kt < NT; ++kt) {
    const int k = kt * 64;
    __syncthreads();
#pragma unroll
    for (int j = 0; j < 4; ++j) {    // 1024 x 16B chunks per tile
      int chunk = t + j * 256;
      int row = chunk >> 3, col = (chunk & 7) * 8;
      *(v8bf*)(&As[0][0] + row * 64 + col) =
          *(const v8bf*)(adjb + (size_t)(n0 + row) * NN + k + col);
      *(v8bf*)(&Bs[0][0] + row * 64 + col) =
          *(const v8bf*)(xpt + (size_t)(c0 + row) * NN + k + col);
    }
    __syncthreads();
    gemm_compute_panel(&As[0][0], &Bs[0][0], wave, r, h, acc);
  }
#endif

#pragma unroll
  for (int tt = 0; tt < 8; ++tt) {
#pragma unroll
    for (int v = 0; v < 8; ++v) {
      int n = n0 + wave * 16 + v + 8 * h;
      int c = c0 + tt * 16 + r;
      cout[(size_t)n * NC + c] = (bf16_t)acc[tt][v];
    }
  }
}

// --- gate GEMM + sigmoid: value[b,o] = sig(sum_i C[n][i*128+b] W0[i][o]+b0) -
__global__ __launch_bounds__(256) void gates_kernel(
    const bf16_t* __restrict__ Cmat, const float* __restrict__ W0,
    const float* __restrict__ b0, const float* __restrict__ state,
    bf16_t* __restrict__ u_ws, bf16_t* __restrict__ rs_ws) {
  __shared__ __align__(16) bf16_t Asl[128 * KPAD];
  __shared__ __align__(16) bf16_t Bsl[128 * KPAD];
  const int t = threadIdx.x;
  const int n = blockIdx.x;
  const int wave = t >> 5, lane = t & 31, r = lane & 15, h = lane >> 4;

  for (int idx = t; idx < (128 * KPAD) / 2; idx += 256) {
    ((unsigned int*)Asl)[idx] = 0u;
    ((unsigned int*)Bsl)[idx] = 0u;
  }
  __syncthreads();
  for (int c = t; c < NC; c += 256)                       // A[b][i] = conv out
    Asl[(c & 127) * KPAD + (c >> 7)] = Cmat[(size_t)n * NC + c];
  for (int idx = t; idx < II * 128; idx += 256)           // B[o][i] = W0^T
    Bsl[(idx & 127) * KPAD + (idx >> 7)] = (bf16_t)W0[idx];
  __syncthreads();

  v8f acc[8] = {};
#pragma unroll
  for (int kc = 0; kc < KPAD; kc += 32) {
    const v16bf a = load_frag(Asl + (wave * 16 + r) * KPAD, kc, h);
    v16bf bfr[8];
#pragma unroll
    for (int tt = 0; tt < 8; ++tt)
      bfr[tt] = load_frag(Bsl + (tt * 16 + r) * KPAD, kc, h);
#pragma unroll
    for (int tt = 0; tt < 8; ++tt)
      acc[tt] = __builtin_amdgcn_wmma_f32_16x16x32_bf16(
          false, a, false, bfr[tt], (short)0, acc[tt], false, false);
  }
#pragma unroll
  for (int tt = 0; tt < 8; ++tt) {
    const int o = tt * 16 + r;
    const float bias = b0[o];
#pragma unroll
    for (int v = 0; v < 8; ++v) {
      int b = wave * 16 + v + 8 * h;
      float g = sigmoid_f(acc[tt][v] + bias);
      size_t base = (size_t)b * NUSZ + (size_t)n * UU;
      if (o < UU) {                       // reset gate -> r * state
        rs_ws[base + o] = (bf16_t)(g * state[base + o]);
      } else {                            // update gate
        u_ws[base + (o - UU)] = (bf16_t)g;
      }
    }
  }
}

// --- candidate GEMM + tanh + GRU combine -----------------------------------
__global__ __launch_bounds__(256) void cand_kernel(
    const bf16_t* __restrict__ Cmat, const float* __restrict__ W1,
    const float* __restrict__ b1, const float* __restrict__ state,
    const bf16_t* __restrict__ u_ws, float* __restrict__ out) {
  __shared__ __align__(16) bf16_t Asl[128 * KPAD];
  __shared__ __align__(16) bf16_t Bsl[64 * KPAD];
  const int t = threadIdx.x;
  const int n = blockIdx.x;
  const int wave = t >> 5, lane = t & 31, r = lane & 15, h = lane >> 4;

  for (int idx = t; idx < (128 * KPAD) / 2; idx += 256)
    ((unsigned int*)Asl)[idx] = 0u;
  for (int idx = t; idx < (64 * KPAD) / 2; idx += 256)
    ((unsigned int*)Bsl)[idx] = 0u;
  __syncthreads();
  for (int c = t; c < NC; c += 256)
    Asl[(c & 127) * KPAD + (c >> 7)] = Cmat[(size_t)n * NC + c];
  for (int idx = t; idx < II * UU; idx += 256)            // B[o][i] = W1^T
    Bsl[(idx & 63) * KPAD + (idx >> 6)] = (bf16_t)W1[idx];
  __syncthreads();

  v8f acc[4] = {};
#pragma unroll
  for (int kc = 0; kc < KPAD; kc += 32) {
    const v16bf a = load_frag(Asl + (wave * 16 + r) * KPAD, kc, h);
    v16bf bfr[4];
#pragma unroll
    for (int tt = 0; tt < 4; ++tt)
      bfr[tt] = load_frag(Bsl + (tt * 16 + r) * KPAD, kc, h);
#pragma unroll
    for (int tt = 0; tt < 4; ++tt)
      acc[tt] = __builtin_amdgcn_wmma_f32_16x16x32_bf16(
          false, a, false, bfr[tt], (short)0, acc[tt], false, false);
  }
#pragma unroll
  for (int tt = 0; tt < 4; ++tt) {
    const int o = tt * 16 + r;
    const float bias = b1[o];
#pragma unroll
    for (int v = 0; v < 8; ++v) {
      int b = wave * 16 + v + 8 * h;
      size_t base = (size_t)b * NUSZ + (size_t)n * UU;
      float c = tanh_f(acc[tt][v] + bias);
      float u = (float)u_ws[base + o];
      float s = state[base + o];
      out[base + o] = u * s + (1.0f - u) * c;
    }
  }
}

// ---------------------------------------------------------------------------
extern "C" void kernel_launch(void* const* d_in, const int* in_sizes, int n_in,
                              void* d_out, int out_size, void* d_ws,
                              size_t ws_size, hipStream_t stream) {
  const float* inputs = (const float*)d_in[0];
  const float* state  = (const float*)d_in[1];
  const float* adj    = (const float*)d_in[2];
  const float* W0     = (const float*)d_in[3];
  const float* b0     = (const float*)d_in[4];
  const float* W1     = (const float*)d_in[5];
  const float* b1     = (const float*)d_in[6];
  float* out = (float*)d_out;

  char* ws = (char*)d_ws;
  bf16_t* adjb  = (bf16_t*)(ws + OFF_ADJB);
  bf16_t* xpt   = (bf16_t*)(ws + OFF_XPT);
  bf16_t* Cm    = (bf16_t*)(ws + OFF_C);
  bf16_t* u_ws  = (bf16_t*)(ws + OFF_U);
  bf16_t* rs_ws = (bf16_t*)(ws + OFF_RS);

  // 1) bf16 packs
  pack_adj_kernel<<<(NN * NN) / (256 * 4), 256, 0, stream>>>(adj, adjb);
  pack_x_f32_kernel<<<dim3(NN / 128, BB), 256, 0, stream>>>(inputs, state, xpt);
  // 2) graph conv #1
  gemm_adj_kernel<<<dim3(NC / 128, NN / 128), 256, 0, stream>>>(adjb, xpt, Cm);
  // 3) gates: sigmoid, u and r*state
  gates_kernel<<<NN, 256, 0, stream>>>(Cm, W0, b0, state, u_ws, rs_ws);
  // 4) repack state columns with r*state
  pack_x_bf16_kernel<<<dim3(NN / 128, BB), 256, 0, stream>>>(rs_ws, xpt);
  // 5) graph conv #2
  gemm_adj_kernel<<<dim3(NC / 128, NN / 128), 256, 0, stream>>>(adjb, xpt, Cm);
  // 6) candidate + GRU combine
  cand_kernel<<<NN, 256, 0, stream>>>(Cm, W1, b1, state, u_ws, out);
}
